// MambaBlock_53403623358853
// MI455X (gfx1250) — matrix operand order repaired and checked by reading, not compiled, not run on hardware
//
#include <hip/hip_runtime.h>
#include <hip/hip_bf16.h>
#include <hip/hip_fp16.h>

// ---------------------------------------------------------------------------
// Mamba block forward for MI455X (gfx1250), wave32 + WMMA f32_16x16x32_f16,
// async global->LDS staging (GLOBAL_LOAD_ASYNC_TO_LDS_B128, ASYNCcnt) with
// double-buffered LDS in the main GEMM.
// Pipeline:
//   0) f32->f16 packs (x, W_in, W_dt, W_out, W_x)
//   1) GEMM1  xr = x @ W_in.T                     (WMMA, f16 out)
//   2) depthwise causal conv(K=4)+bias+SiLU -> xi (f16)
//   3) GEMM2  delta = softplus(xi @ W_dt.T + b)   (WMMA, fused epilogue, f32)
//   4) x_proj Bmat = (xi @ W_x.T)[:,16:32]        (WMMA wave kernel, f32)
//   5) selective scan over SEQ per (b,d) channel  -> yact = (y+u*D)*silu(res), f16
//   6) GEMM3  z = residual + yact @ W_out.T       (WMMA, fused residual, f32)
//   7) LayerNorm(z) -> d_out (f32)
// ---------------------------------------------------------------------------

typedef __attribute__((ext_vector_type(16))) _Float16 v16h;
typedef __attribute__((ext_vector_type(8)))  _Float16 v8h;
typedef __attribute__((ext_vector_type(8)))  float    v8f;

#define BATCH   8
#define SEQ     2048
#define D_MODEL 1024
#define D_STATE 16
#define D_INNER 2048
#define NTOK    (BATCH * SEQ)   // 16384

__device__ __forceinline__ v16h mk16(v8h lo, v8h hi) {
  v16h r;
#pragma unroll
  for (int i = 0; i < 8; ++i) { r[i] = lo[i]; r[8 + i] = hi[i]; }
  return r;
}

__device__ __forceinline__ v8f v8f_zero() {
  v8f z = {0.f, 0.f, 0.f, 0.f, 0.f, 0.f, 0.f, 0.f};
  return z;
}

// 16-byte async copy global -> LDS (VFLAT GLOBAL_LOAD_ASYNC_TO_LDS_B128).
// LDS byte offset = low 32 bits of the generic shared-memory address
// (see ISA 10.2).
__device__ __forceinline__ void async_ld16(const _Float16* gp, _Float16* lp) {
  unsigned loff = (unsigned)(uintptr_t)lp;
  asm volatile("global_load_async_to_lds_b128 %0, %1, off"
               :
               : "v"(loff), "v"(gp)
               : "memory");
}

__device__ __forceinline__ void wait_async0() {
  asm volatile("s_wait_asynccnt 0x0" ::: "memory");
}

// ---------------------------------------------------------------- f32 -> f16
__global__ __launch_bounds__(256) void cvt_f16_kernel(const float* __restrict__ in,
                                                      _Float16* __restrict__ out, int n) {
  int i = blockIdx.x * 256 + threadIdx.x;
  if (i < n) out[i] = (_Float16)in[i];
}

// ------------------------------------------------------------ main WMMA GEMM
// C[M,N] = A[M,K] @ W[N,K]^T.  A,W f16 K-contiguous.  M=16384 implicit.
// Block: 256 thr (8 waves), tile 128(M) x 64(N), K-step 32, double-buffered
// async LDS staging.
// EPI 0: store f16.  EPI 1: softplus(acc + aux[col]) -> f32.
// EPI 2: acc + aux[row*N+col] (residual) -> f32.
template <int EPI>
__global__ __launch_bounds__(256)
void gemm_wmma_kernel(const _Float16* __restrict__ A, const _Float16* __restrict__ W,
                      int K, int N, void* __restrict__ Cout,
                      const float* __restrict__ aux) {
  __shared__ alignas(16) _Float16 As[2][128][40];   // pad 32 -> 40 halves
  __shared__ alignas(16) _Float16 Bs[2][64][40];

  const int tid  = threadIdx.x;
  const int lane = tid & 31;
  const int wv   = tid >> 5;      // 0..7
  const int wm   = wv & 3;        // M quadrant (32 rows each)
  const int wn   = wv >> 2;       // N half (32 cols each)
  const int ln   = lane & 15;
  const int hi   = lane >> 4;     // lane half select
  const int m0   = blockIdx.y * 128;
  const int n0   = blockIdx.x * 64;

  // per-thread staging assignments (fixed): A: 2 x 16B chunks, W: 1 x 16B
  const int ar0 = (tid * 2) >> 2,     aq0 = (tid * 2) & 3;
  const int ar1 = (tid * 2 + 1) >> 2, aq1 = (tid * 2 + 1) & 3;
  const int br  = tid >> 2,           bq  = tid & 3;

  auto stage = [&](int buf, int k0) {
    async_ld16(A + (size_t)(m0 + ar0) * K + k0 + aq0 * 8, &As[buf][ar0][aq0 * 8]);
    async_ld16(A + (size_t)(m0 + ar1) * K + k0 + aq1 * 8, &As[buf][ar1][aq1 * 8]);
    async_ld16(W + (size_t)(n0 + br) * K + k0 + bq * 8,   &Bs[buf][br][bq * 8]);
  };

  v8f acc[2][2];
#pragma unroll
  for (int i = 0; i < 2; ++i)
#pragma unroll
    for (int j = 0; j < 2; ++j) acc[i][j] = v8f_zero();

  stage(0, 0);
  int buf = 0;
  for (int k0 = 0; k0 < K; k0 += 32) {
    wait_async0();        // our async writes for `buf` complete
    __syncthreads();      // everyone's async writes for `buf` complete
    if (k0 + 32 < K) stage(buf ^ 1, k0 + 32);   // overlap next tile with compute

    // ---- fragments (ISA 7.12.2 layouts)
    v16h af[2], bf[2];
#pragma unroll
    for (int im = 0; im < 2; ++im) {
      const _Float16* p = &As[buf][wm * 32 + im * 16 + ln][0];
      af[im] = mk16(*(const v8h*)(p + hi * 8), *(const v8h*)(p + 16 + hi * 8));
    }
#pragma unroll
    for (int jn = 0; jn < 2; ++jn) {
      const _Float16* p = &Bs[buf][wn * 32 + jn * 16 + ln][0];
      bf[jn] = mk16(*(const v8h*)(p + hi * 8), *(const v8h*)(p + 16 + hi * 8));
    }

#pragma unroll
    for (int im = 0; im < 2; ++im)
#pragma unroll
      for (int jn = 0; jn < 2; ++jn)
        acc[im][jn] = __builtin_amdgcn_wmma_f32_16x16x32_f16(
            false, af[im], false, bf[jn], (short)0, acc[im][jn], false, false);
    buf ^= 1;
  }

  // ---- epilogue: VGPR r -> row r + hi*8, col = ln
#pragma unroll
  for (int im = 0; im < 2; ++im) {
#pragma unroll
    for (int jn = 0; jn < 2; ++jn) {
      int rowb = m0 + wm * 32 + im * 16 + hi * 8;
      int col  = n0 + wn * 32 + jn * 16 + ln;
#pragma unroll
      for (int r = 0; r < 8; ++r) {
        size_t off = (size_t)(rowb + r) * N + col;
        float v = acc[im][jn][r];
        if (EPI == 0) {
          ((_Float16*)Cout)[off] = (_Float16)v;
        } else if (EPI == 1) {
          float s  = v + aux[col];
          float sp = (s > 20.f) ? s : log1pf(expf(s));
          ((float*)Cout)[off] = sp;
        } else {
          ((float*)Cout)[off] = v + aux[off];
        }
      }
    }
  }
}

// ------------------------------------------- depthwise causal conv + SiLU
// xr layout: [tok, 4096] (first 2048 = xi_pre, last 2048 = res). Output xih f16.
__global__ __launch_bounds__(256)
void conv_silu_kernel(const _Float16* __restrict__ xr, const float* __restrict__ cw,
                      const float* __restrict__ cb, _Float16* __restrict__ xih) {
  const int d  = blockIdx.x * 256 + threadIdx.x;   // 0..2047
  const int b  = blockIdx.y;
  const int l0 = blockIdx.z * 128;
  const float w0 = cw[d * 4 + 0], w1 = cw[d * 4 + 1], w2 = cw[d * 4 + 2], w3 = cw[d * 4 + 3];
  const float bb = cb[d];

  auto ld = [&](int l) -> float {
    return (l < 0) ? 0.f : (float)xr[(size_t)(b * SEQ + l) * 4096 + d];
  };
  float xm3 = ld(l0 - 3), xm2 = ld(l0 - 2), xm1 = ld(l0 - 1);
  for (int li = 0; li < 128; ++li) {
    int l = l0 + li;
    float x0 = ld(l);
    float v  = w0 * xm3 + w1 * xm2 + w2 * xm1 + w3 * x0 + bb;
    float s  = v / (1.f + expf(-v));   // SiLU
    xih[(size_t)(b * SEQ + l) * D_INNER + d] = (_Float16)s;
    xm3 = xm2; xm2 = xm1; xm1 = x0;
  }
}

// --------------------------------------- x_proj: Bmat = (xi @ W_x.T)[:,16:32]
// One wave per 16-row M tile, single 16-wide N tile, direct global fragments.
__global__ __launch_bounds__(256)
void xproj_wmma_kernel(const _Float16* __restrict__ xih, const _Float16* __restrict__ wxh,
                       float* __restrict__ Bmat) {
  const int lane  = threadIdx.x & 31;
  const int wv    = threadIdx.x >> 5;
  const int mbase = (blockIdx.x * 8 + wv) * 16;
  const int ln    = lane & 15;
  const int hi    = lane >> 4;

  v8f acc = v8f_zero();
  const _Float16* arow = xih + (size_t)(mbase + ln) * D_INNER;
  const _Float16* brow = wxh + (size_t)(16 + ln) * D_INNER;   // rows 16..31 of W_x
  for (int k0 = 0; k0 < D_INNER; k0 += 32) {
    v16h a = mk16(*(const v8h*)(arow + k0 + hi * 8),
                  *(const v8h*)(arow + k0 + 16 + hi * 8));
    v16h b = mk16(*(const v8h*)(brow + k0 + hi * 8),
                  *(const v8h*)(brow + k0 + 16 + hi * 8));
    acc = __builtin_amdgcn_wmma_f32_16x16x32_f16(false, a, false, b, (short)0, acc,
                                                 false, false);
  }
#pragma unroll
  for (int r = 0; r < 8; ++r)
    Bmat[(size_t)(mbase + r + hi * 8) * D_STATE + ln] = acc[r];
}

// ------------------------------------------------------- selective scan
// One thread per (b,d) channel, h[16] in registers, B_t staged via LDS.
__global__ __launch_bounds__(256)
void scan_kernel(const float* __restrict__ delta, const _Float16* __restrict__ xih,
                 const _Float16* __restrict__ xr, const float* __restrict__ Bmat,
                 const float* __restrict__ A_log, const float* __restrict__ Dskip,
                 _Float16* __restrict__ yact) {
  __shared__ float Bsh[128 * D_STATE];   // 8 KB
  const int c = blockIdx.x * 256 + threadIdx.x;
  const int b = c >> 11;          // D_INNER == 2048
  const int d = c & (D_INNER - 1);

  float a[D_STATE], h[D_STATE];
#pragma unroll
  for (int n = 0; n < D_STATE; ++n) {
    a[n] = -expf(A_log[(size_t)d * D_STATE + n]);
    h[n] = 0.f;
  }
  const float dsk = Dskip[d];

  for (int lc = 0; lc < SEQ; lc += 128) {
    __syncthreads();
    for (int i = threadIdx.x; i < 128 * D_STATE; i += 256) {
      int ll = i >> 4, nn = i & 15;
      Bsh[i] = Bmat[(size_t)(b * SEQ + lc + ll) * D_STATE + nn];
    }
    __syncthreads();
    for (int li = 0; li < 128; ++li) {
      const size_t t  = (size_t)b * SEQ + lc + li;
      const float  dt = delta[t * D_INNER + d];
      const float  u  = (float)xih[t * D_INNER + d];
      const float  rv = (float)xr[t * 4096 + D_INNER + d];   // res half
      const float* Bl = &Bsh[li * D_STATE];
      const float  du = dt * u;
      float y = 0.f;
#pragma unroll
      for (int n = 0; n < D_STATE; ++n) {
        h[n] = expf(dt * a[n]) * h[n] + du * Bl[n];
        y   += h[n] * Bl[n];
      }
      float silu_r = rv / (1.f + expf(-rv));
      yact[t * D_INNER + d] = (_Float16)((y + u * dsk) * silu_r);
    }
  }
}

// ------------------------------------------------------------ LayerNorm
__global__ __launch_bounds__(256)
void ln_kernel(const float* __restrict__ z, const float* __restrict__ g,
               const float* __restrict__ be, float* __restrict__ out) {
  __shared__ float s1[256], s2[256];
  const size_t t = blockIdx.x;
  const float* zr = z + t * D_MODEL;
  float v[4], sum = 0.f, sq = 0.f;
#pragma unroll
  for (int k = 0; k < 4; ++k) {
    v[k] = zr[threadIdx.x + k * 256];
    sum += v[k]; sq += v[k] * v[k];
  }
  s1[threadIdx.x] = sum; s2[threadIdx.x] = sq;
  __syncthreads();
  for (int o = 128; o > 0; o >>= 1) {
    if (threadIdx.x < o) {
      s1[threadIdx.x] += s1[threadIdx.x + o];
      s2[threadIdx.x] += s2[threadIdx.x + o];
    }
    __syncthreads();
  }
  const float mu  = s1[0] * (1.f / D_MODEL);
  const float var = s2[0] * (1.f / D_MODEL) - mu * mu;
  const float rs  = rsqrtf(var + 1e-5f);
#pragma unroll
  for (int k = 0; k < 4; ++k) {
    int col = threadIdx.x + k * 256;
    out[t * D_MODEL + col] = (v[k] - mu) * rs * g[col] + be[col];
  }
}

// ---------------------------------------------------------------------------
extern "C" void kernel_launch(void* const* d_in, const int* in_sizes, int n_in,
                              void* d_out, int out_size, void* d_ws, size_t ws_size,
                              hipStream_t stream) {
  const float* x      = (const float*)d_in[0];
  const float* W_in   = (const float*)d_in[1];
  const float* conv_w = (const float*)d_in[2];
  const float* conv_b = (const float*)d_in[3];
  const float* W_x    = (const float*)d_in[4];
  const float* W_dt   = (const float*)d_in[5];
  const float* b_dt   = (const float*)d_in[6];
  const float* A_log  = (const float*)d_in[7];
  const float* D_skip = (const float*)d_in[8];
  const float* W_out  = (const float*)d_in[9];
  const float* ln_g   = (const float*)d_in[10];
  const float* ln_b   = (const float*)d_in[11];

  char* ws = (char*)d_ws;
  // workspace layout (bytes)
  constexpr size_t o_xh    = 0;                                        // 32 MB f16 x
  constexpr size_t o_win   = o_xh    + (size_t)NTOK * D_MODEL * 2;     // 8 MB
  constexpr size_t o_wdt   = o_win   + (size_t)2 * D_INNER * D_MODEL * 2;  // 8 MB
  constexpr size_t o_wout  = o_wdt   + (size_t)D_INNER * D_INNER * 2;  // 4 MB
  constexpr size_t o_wx    = o_wout  + (size_t)D_MODEL * D_INNER * 2;  // 128 KB
  constexpr size_t o_xr    = o_wx    + (size_t)2 * D_STATE * D_INNER * 2;  // 128 MB f16
  constexpr size_t o_xih   = o_xr    + (size_t)NTOK * 2 * D_INNER * 2; // 64 MB f16
  constexpr size_t o_delta = o_xih   + (size_t)NTOK * D_INNER * 2;     // 128 MB f32
  constexpr size_t o_bmat  = o_delta + (size_t)NTOK * D_INNER * 4;     // 1 MB f32
  constexpr size_t o_yact  = o_bmat  + (size_t)NTOK * D_STATE * 4;     // 64 MB f16
  constexpr size_t o_z     = o_yact  + (size_t)NTOK * D_INNER * 2;     // 64 MB f32

  _Float16* xh    = (_Float16*)(ws + o_xh);
  _Float16* winh  = (_Float16*)(ws + o_win);
  _Float16* wdth  = (_Float16*)(ws + o_wdt);
  _Float16* wouth = (_Float16*)(ws + o_wout);
  _Float16* wxh   = (_Float16*)(ws + o_wx);
  _Float16* xr    = (_Float16*)(ws + o_xr);
  _Float16* xih   = (_Float16*)(ws + o_xih);
  float*    delta = (float*)   (ws + o_delta);
  float*    bmat  = (float*)   (ws + o_bmat);
  _Float16* yact  = (_Float16*)(ws + o_yact);
  float*    zbuf  = (float*)   (ws + o_z);

  auto cvt = [&](const float* src, _Float16* dst, int n) {
    cvt_f16_kernel<<<(n + 255) / 256, 256, 0, stream>>>(src, dst, n);
  };
  // 0) packs
  cvt(x,     xh,    NTOK * D_MODEL);
  cvt(W_in,  winh,  2 * D_INNER * D_MODEL);
  cvt(W_dt,  wdth,  D_INNER * D_INNER);
  cvt(W_out, wouth, D_MODEL * D_INNER);
  cvt(W_x,   wxh,   2 * D_STATE * D_INNER);

  // 1) xr = x @ W_in.T   [16384,1024]x[4096,1024]^T
  gemm_wmma_kernel<0><<<dim3(4096 / 64, NTOK / 128), 256, 0, stream>>>(
      xh, winh, D_MODEL, 2 * D_INNER, (void*)xr, nullptr);

  // 2) conv + SiLU -> xih
  conv_silu_kernel<<<dim3(D_INNER / 256, BATCH, SEQ / 128), 256, 0, stream>>>(
      xr, conv_w, conv_b, xih);

  // 3) delta = softplus(xi @ W_dt.T + b_dt)
  gemm_wmma_kernel<1><<<dim3(D_INNER / 64, NTOK / 128), 256, 0, stream>>>(
      xih, wdth, D_INNER, D_INNER, (void*)delta, b_dt);

  // 4) Bmat = (xi @ W_x.T)[:,16:32]
  xproj_wmma_kernel<<<NTOK / 16 / 8, 256, 0, stream>>>(xih, wxh, bmat);

  // 5) selective scan -> yact
  scan_kernel<<<(BATCH * D_INNER) / 256, 256, 0, stream>>>(
      delta, xih, xr, bmat, A_log, D_skip, yact);

  // 6) z = x + yact @ W_out.T
  gemm_wmma_kernel<2><<<dim3(D_MODEL / 64, NTOK / 128), 256, 0, stream>>>(
      yact, wouth, D_INNER, D_MODEL, (void*)zbuf, x);

  // 7) LayerNorm -> d_out
  ln_kernel<<<NTOK, 256, 0, stream>>>(zbuf, ln_g, ln_b, (float*)d_out);
}
